// UMC_Loss_67190468379034
// MI455X (gfx1250) — matrix operand order repaired and checked by reading, not compiled, not run on hardware
//
#include <hip/hip_runtime.h>
#include <hip/hip_bf16.h>
#include <math.h>

// ---------------------------------------------------------------------------
// Problem constants (reference: N=M=8192, D=128, B=1, eps=tau=1, 50 iters)
// ---------------------------------------------------------------------------
#define NP    8192
#define DIMF  128
#define MAXIT 50
#define TOLC  1e-5f
#define ALPHA (1.0f / 8192.0f)

typedef __attribute__((ext_vector_type(16))) __bf16        v16bf;
typedef __attribute__((ext_vector_type(8)))  float         v8f;
typedef __attribute__((ext_vector_type(8)))  unsigned int  v8u;

// ---------------------------------------------------------------------------
// bf16 hi/lo split helpers (S = Ahi*Bhi + Ahi*Blo + Alo*Bhi ~ fp32 accuracy)
// ---------------------------------------------------------------------------
__device__ __forceinline__ unsigned pk2(float a, float b) {
  unsigned ua = __builtin_bit_cast(unsigned, a);
  unsigned ub = __builtin_bit_cast(unsigned, b);
  return (ua >> 16) | (ub & 0xFFFF0000u);
}
__device__ __forceinline__ float tbf(float a) {
  return __builtin_bit_cast(float, __builtin_bit_cast(unsigned, a) & 0xFFFF0000u);
}

struct BfPair { v16bf hi, lo; };

__device__ __forceinline__ BfPair mk_bfpair(float4 a, float4 b, float4 c, float4 d) {
  float f[16] = {a.x,a.y,a.z,a.w, b.x,b.y,b.z,b.w, c.x,c.y,c.z,c.w, d.x,d.y,d.z,d.w};
  v8u uh, ul;
#pragma unroll
  for (int i = 0; i < 8; ++i) {
    float x = f[2*i], y = f[2*i+1];
    float xh = tbf(x), yh = tbf(y);
    uh[i] = pk2(xh, yh);
    ul[i] = pk2(x - xh, y - yh);
  }
  BfPair r;
  r.hi = __builtin_bit_cast(v16bf, uh);
  r.lo = __builtin_bit_cast(v16bf, ul);
  return r;
}

// A-matrix 16x32 bf16 layout: lane l -> row tileM+(l&15); h=l>>4;
// elements 0..7 = K[kb+8h .. +7], elements 8..15 = K[kb+16+8h .. +7]
__device__ __forceinline__ BfPair load_A(const float* base, int row, int kb, int h) {
  const float4* q = (const float4*)(base + (size_t)row * DIMF + kb + 8 * h);
  return mk_bfpair(q[0], q[1], q[4], q[5]);
}
// B-matrix 32x16 bf16 layout: lane l -> col tileN+(l&15); h=l>>4;
// elements 0..7 = K[kb+8h .. +7], elements 8..15 = K[kb+16+8h .. +7]
__device__ __forceinline__ BfPair load_B(const float* base, int row, int kb, int h) {
  const float4* q = (const float4*)(base + (size_t)row * DIMF + kb + 8 * h);
  return mk_bfpair(q[0], q[1], q[4], q[5]);
}

// ---------------------------------------------------------------------------
// Block reductions (fixed order -> deterministic)
// ---------------------------------------------------------------------------
template <int BS>
__device__ __forceinline__ float blk_sum(float v, float* sm) {
  int t = threadIdx.x;
  sm[t] = v; __syncthreads();
#pragma unroll
  for (int s = BS / 2; s > 0; s >>= 1) {
    if (t < s) sm[t] += sm[t + s];
    __syncthreads();
  }
  float r = sm[0]; __syncthreads();
  return r;
}
template <int BS>
__device__ __forceinline__ float blk_max(float v, float* sm) {
  int t = threadIdx.x;
  sm[t] = v; __syncthreads();
#pragma unroll
  for (int s = BS / 2; s > 0; s >>= 1) {
    if (t < s) sm[t] = fmaxf(sm[t], sm[t + s]);
    __syncthreads();
  }
  float r = sm[0]; __syncthreads();
  return r;
}

// ---------------------------------------------------------------------------
// Row squared norms: one wave per row (32 lanes x float4 = 128 elems)
// ---------------------------------------------------------------------------
__global__ void __launch_bounds__(256) norms_kernel(const float* __restrict__ f,
                                                    float* __restrict__ out) {
  int row  = blockIdx.x * 8 + (threadIdx.x >> 5);
  int lane = threadIdx.x & 31;
  float4 x = ((const float4*)(f + (size_t)row * DIMF))[lane];
  float s = x.x * x.x + x.y * x.y + x.z * x.z + x.w * x.w;
#pragma unroll
  for (int o = 16; o > 0; o >>= 1) s += __shfl_xor(s, o, 32);
  if (lane == 0) out[row] = s;
}

// ---------------------------------------------------------------------------
// Build K = exp(-sqrt(max(|q|^2 + |k|^2 - 2 q.k, 0)))  (eps = 1)
// One wave per 16x16 tile; 12 v_wmma_f32_16x16x32_bf16 per tile.
// ---------------------------------------------------------------------------
__global__ void __launch_bounds__(256) build_K_kernel(const float* __restrict__ Q,
                                                      const float* __restrict__ F,
                                                      const float* __restrict__ qn,
                                                      const float* __restrict__ kn,
                                                      float* __restrict__ Kout) {
  int lane  = threadIdx.x & 31;
  int wave  = threadIdx.x >> 5;
  int tileM = blockIdx.y * 16;
  int tileN = (blockIdx.x * 8 + wave) * 16;
  int h     = lane >> 4;
  int l15   = lane & 15;
  int am    = tileM + l15;
  int bn    = tileN + l15;

  v8f acc = {0.f, 0.f, 0.f, 0.f, 0.f, 0.f, 0.f, 0.f};
#pragma unroll
  for (int kb = 0; kb < DIMF; kb += 32) {
    BfPair A = load_A(Q, am, kb, h);
    BfPair B = load_B(F, bn, kb, h);
    acc = __builtin_amdgcn_wmma_f32_16x16x32_bf16(false, A.hi, false, B.hi,
                                                  (short)0, acc, false, false);
    acc = __builtin_amdgcn_wmma_f32_16x16x32_bf16(false, A.hi, false, B.lo,
                                                  (short)0, acc, false, false);
    acc = __builtin_amdgcn_wmma_f32_16x16x32_bf16(false, A.lo, false, B.hi,
                                                  (short)0, acc, false, false);
  }

  int   n   = tileN + l15;
  float knv = kn[n];
#pragma unroll
  for (int r = 0; r < 8; ++r) {
    int   m  = tileM + r + 8 * h;           // C/D layout: VGPR r <-> row r+8h
    float s  = acc[r];
    float d2 = fmaf(-2.f, s, qn[m] + knv);
    float d  = sqrtf(fmaxf(d2, 0.f));
    Kout[(size_t)m * NP + n] = expf(-d);
  }
}

// ---------------------------------------------------------------------------
// Kv[i] = sum_j K[i,j] * v[j]  -- one row per block, streaming float4
// ---------------------------------------------------------------------------
__global__ void __launch_bounds__(256) kv_kernel(const float* __restrict__ K,
                                                 const float* __restrict__ v,
                                                 float* __restrict__ Kv,
                                                 const int* __restrict__ done) {
  if (*done) return;
  __shared__ float sm[256];
  const float* row = K + (size_t)blockIdx.x * NP;
  float acc = 0.f;
  for (int j = threadIdx.x * 4; j < NP; j += 1024) {
    float4 a = *(const float4*)(row + j);
    float4 b = *(const float4*)(v + j);
    acc = fmaf(a.x, b.x, acc); acc = fmaf(a.y, b.y, acc);
    acc = fmaf(a.z, b.z, acc); acc = fmaf(a.w, b.w, acc);
  }
  float s = blk_sum<256>(acc, sm);
  if (threadIdx.x == 0) Kv[blockIdx.x] = s;
}

// ---------------------------------------------------------------------------
// uK[j] = sum_i u[i] * K[i,j]  -- 32 row-chunks x 32 col-blocks, then reduce
// ---------------------------------------------------------------------------
__global__ void __launch_bounds__(256) uk_partial_kernel(const float* __restrict__ K,
                                                         const float* __restrict__ u,
                                                         float* __restrict__ part,
                                                         const int* __restrict__ done) {
  if (*done) return;
  int j  = blockIdx.x * 256 + threadIdx.x;
  int i0 = blockIdx.y * 256;
  float acc = 0.f;
  for (int i = i0; i < i0 + 256; ++i)
    acc = fmaf(u[i], K[(size_t)i * NP + j], acc);
  part[(size_t)blockIdx.y * NP + j] = acc;
}

__global__ void __launch_bounds__(256) uk_reduce_kernel(const float* __restrict__ part,
                                                        float* __restrict__ uK,
                                                        const int* __restrict__ done) {
  if (*done) return;
  int j = blockIdx.x * 256 + threadIdx.x;
  float acc = 0.f;
#pragma unroll
  for (int c = 0; c < 32; ++c) acc += part[(size_t)c * NP + j];
  uK[j] = acc;
}

// ---------------------------------------------------------------------------
// u_new = (alpha - log(Kv + 1)) - logsumexp(.) ; md_u = max|u_new - u|
// ---------------------------------------------------------------------------
__global__ void __launch_bounds__(1024) u_update_kernel(const float* __restrict__ Kv,
                                                        const float* __restrict__ u,
                                                        float* __restrict__ u_new,
                                                        float* __restrict__ md_u,
                                                        const int* __restrict__ done) {
  if (*done) return;
  __shared__ float sm[1024];
  int tid = threadIdx.x;
  float t[8];
  float lmax = -1e30f;
#pragma unroll
  for (int e = 0; e < 8; ++e) {
    int i = tid + e * 1024;
    float x = ALPHA - logf(Kv[i] + 1.0f);   // tau = eps = 1
    t[e] = x; lmax = fmaxf(lmax, x);
  }
  float m = blk_max<1024>(lmax, sm);
  float ls = 0.f;
#pragma unroll
  for (int e = 0; e < 8; ++e) ls += expf(t[e] - m);
  float lse = m + logf(blk_sum<1024>(ls, sm));
  float dmax = 0.f;
#pragma unroll
  for (int e = 0; e < 8; ++e) {
    int i = tid + e * 1024;
    float un = t[e] - lse;
    u_new[i] = un;
    dmax = fmaxf(dmax, fabsf(un - u[i]));
  }
  float md = blk_max<1024>(dmax, sm);
  if (tid == 0) md_u[0] = md;
}

// ---------------------------------------------------------------------------
// v_new + convergence check + freeze/commit (reference freezes BEFORE assign)
// ---------------------------------------------------------------------------
__global__ void __launch_bounds__(1024) v_update_kernel(const float* __restrict__ uK,
                                                        float* __restrict__ u,
                                                        float* __restrict__ v,
                                                        const float* __restrict__ u_new,
                                                        const float* __restrict__ md_u,
                                                        int* __restrict__ done) {
  if (*done) return;
  __shared__ float sm[1024];
  __shared__ int s_commit;
  int tid = threadIdx.x;
  float t[8];
  float lmax = -1e30f;
#pragma unroll
  for (int e = 0; e < 8; ++e) {
    int i = tid + e * 1024;
    float x = ALPHA - logf(uK[i] + 1.0f);
    t[e] = x; lmax = fmaxf(lmax, x);
  }
  float m = blk_max<1024>(lmax, sm);
  float ls = 0.f;
#pragma unroll
  for (int e = 0; e < 8; ++e) ls += expf(t[e] - m);
  float lse = m + logf(blk_sum<1024>(ls, sm));
  float vn[8];
  float dmax = 0.f;
#pragma unroll
  for (int e = 0; e < 8; ++e) {
    int i = tid + e * 1024;
    vn[e] = t[e] - lse;
    dmax = fmaxf(dmax, fabsf(vn[e] - v[i]));
  }
  float mdv = blk_max<1024>(dmax, sm);
  if (tid == 0) {
    bool conv = (mdv < TOLC) && (md_u[0] < TOLC);
    if (conv) { *done = 1; s_commit = 0; } else { s_commit = 1; }
  }
  __syncthreads();
  if (s_commit) {
#pragma unroll
    for (int e = 0; e < 8; ++e) {
      int i = tid + e * 1024;
      v[i] = vn[e];
      u[i] = u_new[i];
    }
  }
}

__global__ void init_kernel(float* u, float* v, int* done, float* md_u) {
  int i = blockIdx.x * 256 + threadIdx.x;
  if (i < NP) { u[i] = 0.f; v[i] = 0.f; }
  if (i == 0) { *done = 0; md_u[0] = 0.f; }
}

// ---------------------------------------------------------------------------
// rowpart[i] = sum_j |u1_i v1_j K1_ij - u2_i v2_j K2_ij| ; then final mean
// ---------------------------------------------------------------------------
__global__ void __launch_bounds__(256) rowloss_kernel(const float* __restrict__ K1,
                                                      const float* __restrict__ u1,
                                                      const float* __restrict__ v1,
                                                      const float* __restrict__ K2,
                                                      const float* __restrict__ u2,
                                                      const float* __restrict__ v2,
                                                      float* __restrict__ rowpart) {
  __shared__ float sm[256];
  int i = blockIdx.x;
  float a = u1[i], b = u2[i];
  const float* r1 = K1 + (size_t)i * NP;
  const float* r2 = K2 + (size_t)i * NP;
  float acc = 0.f;
  for (int j = threadIdx.x * 4; j < NP; j += 1024) {
    float4 k1 = *(const float4*)(r1 + j), k2 = *(const float4*)(r2 + j);
    float4 x1 = *(const float4*)(v1 + j), x2 = *(const float4*)(v2 + j);
    acc += fabsf(a * x1.x * k1.x - b * x2.x * k2.x);
    acc += fabsf(a * x1.y * k1.y - b * x2.y * k2.y);
    acc += fabsf(a * x1.z * k1.z - b * x2.z * k2.z);
    acc += fabsf(a * x1.w * k1.w - b * x2.w * k2.w);
  }
  float s = blk_sum<256>(acc, sm);
  if (threadIdx.x == 0) rowpart[i] = s;
}

__global__ void __launch_bounds__(1024) final_kernel(const float* __restrict__ rowpart,
                                                     float* __restrict__ out) {
  __shared__ float sm[1024];
  float acc = 0.f;
#pragma unroll
  for (int e = 0; e < 8; ++e) acc += rowpart[threadIdx.x + e * 1024];
  float s = blk_sum<1024>(acc, sm);
  if (threadIdx.x == 0) out[0] = s * (1.0f / ((float)NP * (float)NP));
}

// ---------------------------------------------------------------------------
// Orchestration
// ---------------------------------------------------------------------------
extern "C" void kernel_launch(void* const* d_in, const int* in_sizes, int n_in,
                              void* d_out, int out_size, void* d_ws, size_t ws_size,
                              hipStream_t stream) {
  (void)in_sizes; (void)n_in; (void)out_size; (void)ws_size;
  const float* src = (const float*)d_in[0];
  const float* tgt = (const float*)d_in[1];
  const float* gen = (const float*)d_in[2];
  float* out = (float*)d_out;

  char* ws = (char*)d_ws;
  size_t off = 0;
  auto alloc = [&](size_t bytes) -> void* {
    void* p = ws + off;
    off = (off + bytes + 255) & ~(size_t)255;
    return p;
  };
  const size_t KB_BYTES = (size_t)NP * NP * sizeof(float);   // 256 MB
  float* K1     = (float*)alloc(KB_BYTES);
  float* K2     = (float*)alloc(KB_BYTES);
  float* n_src  = (float*)alloc(NP * sizeof(float));
  float* n_tgt  = (float*)alloc(NP * sizeof(float));
  float* n_gen  = (float*)alloc(NP * sizeof(float));
  float* Kv     = (float*)alloc(NP * sizeof(float));
  float* uK     = (float*)alloc(NP * sizeof(float));
  float* ukpart = (float*)alloc((size_t)32 * NP * sizeof(float));
  float* u1     = (float*)alloc(NP * sizeof(float));
  float* v1     = (float*)alloc(NP * sizeof(float));
  float* u2     = (float*)alloc(NP * sizeof(float));
  float* v2     = (float*)alloc(NP * sizeof(float));
  float* u_new  = (float*)alloc(NP * sizeof(float));
  float* rowpart= (float*)alloc(NP * sizeof(float));
  int*   done   = (int*)alloc(256);
  float* md_u   = (float*)alloc(256);

  // Row norms for all three feature sets
  norms_kernel<<<NP / 8, 256, 0, stream>>>(src, n_src);
  norms_kernel<<<NP / 8, 256, 0, stream>>>(tgt, n_tgt);
  norms_kernel<<<NP / 8, 256, 0, stream>>>(gen, n_gen);

  // K1 = Gibbs(src, tgt), K2 = Gibbs(tgt, gen)
  dim3 gK(NP / (16 * 8), NP / 16);   // (64, 512): 8 waves/block, 1 tile/wave
  build_K_kernel<<<gK, 256, 0, stream>>>(src, tgt, n_src, n_tgt, K1);
  build_K_kernel<<<gK, 256, 0, stream>>>(tgt, gen, n_tgt, n_gen, K2);

  auto sinkhorn = [&](const float* K, float* u, float* v) {
    init_kernel<<<NP / 256, 256, 0, stream>>>(u, v, done, md_u);
    for (int it = 0; it < MAXIT; ++it) {
      kv_kernel<<<NP, 256, 0, stream>>>(K, v, Kv, done);
      u_update_kernel<<<1, 1024, 0, stream>>>(Kv, u, u_new, md_u, done);
      uk_partial_kernel<<<dim3(NP / 256, 32), 256, 0, stream>>>(K, u_new, ukpart, done);
      uk_reduce_kernel<<<NP / 256, 256, 0, stream>>>(ukpart, uK, done);
      v_update_kernel<<<1, 1024, 0, stream>>>(uK, u, v, u_new, md_u, done);
    }
  };
  sinkhorn(K1, u1, v1);
  sinkhorn(K2, u2, v2);

  rowloss_kernel<<<NP, 256, 0, stream>>>(K1, u1, v1, K2, u2, v2, rowpart);
  final_kernel<<<1, 1024, 0, stream>>>(rowpart, out);
}
